// MMs_9698036155138
// MI455X (gfx1250) — matrix operand order repaired and checked by reading, not compiled, not run on hardware
//
#include <hip/hip_runtime.h>

// ---------------- problem constants ----------------
#define BATCH   16
#define SEQL    2048
#define DMODEL  256
#define DSTATE  16
#define DINNER  512
#define DTRANK  16
#define NOUT    256
#define BL      (BATCH * SEQL)      // 32768 tokens
#define XDBL_P  64                  // x_dbl padded cols (48 -> 64)
#define CHUNK   128                 // scan chunk length
#define NCHUNK  (SEQL / CHUNK)      // 16
#define LN_EPS  1e-5f

typedef __attribute__((ext_vector_type(16))) __bf16 v16bf;
typedef __attribute__((ext_vector_type(8)))  float  v8f;
typedef __attribute__((ext_vector_type(4)))  unsigned int u32x4;

union Frag16 { v16bf v; u32x4 q[2]; };

__device__ __forceinline__ float sigf(float x) { return 1.0f / (1.0f + __expf(-x)); }

// ---------------- elementwise prep ----------------
__global__ void k_cvt_f32_bf16(const float* __restrict__ src, __bf16* __restrict__ dst, int n) {
    int i = blockIdx.x * 256 + threadIdx.x;
    if (i < n) dst[i] = (__bf16)src[i];
}

// W (N,K) f32 row-major  ->  Wt (K,Npad) bf16 row-major, zero padded
__global__ void k_transpose_cvt(const float* __restrict__ src, __bf16* __restrict__ dst,
                                int N, int K, int Npad) {
    int i = blockIdx.x * 256 + threadIdx.x;
    int total = K * Npad;
    if (i >= total) return;
    int k = i / Npad, n = i % Npad;
    dst[i] = (n < N) ? (__bf16)src[(long)n * K + k] : (__bf16)0.0f;
}

// ---------------- bf16 WMMA GEMM: C(M,N) = A(M,K) @ Wt(K,N) ----------------
// Block tile 128x64 (8 waves, each wave a 32x32 = 2x2 fragment tile).
// Register-pipelined global->LDS staging (next tile loads overlap WMMA work).
// EPI 0: store f32   | EPI 1: silu -> bf16 store | EPI 2: +bias -> f32 store
template <int EPI>
__global__ __launch_bounds__(256)
void gemm_bf16(const __bf16* __restrict__ A, const __bf16* __restrict__ Wt,
               float* __restrict__ outF, __bf16* __restrict__ outH,
               const float* __restrict__ bias, int M, int N, int K) {
    __shared__ __bf16 sA[128 * 40];  // M x K tile, stride 40 (80B, 16B aligned)
    __shared__ __bf16 sB[32 * 72];   // K x N tile, stride 72 (144B, 16B aligned)

    const int tid  = threadIdx.x;
    const int lane = tid & 31;
    const int w    = tid >> 5;       // 8 waves
    const int mw   = w >> 1;         // 0..3 (M sub-tile of 32)
    const int nw   = w & 1;          // 0..1 (N sub-tile of 32)
    const int half = lane >> 4;      // K-half select for A frag / M-half for C
    const int r16  = lane & 15;

    const long m0 = (long)blockIdx.x * 128;
    const long n0 = (long)blockIdx.y * 64;

    // global->LDS copy assignments
    const int arow = tid >> 2, achk = tid & 3;   // A: rows [arow, arow+64], 8-bf16 chunks
    const int brow = tid >> 3, bchk = tid & 7;   // B: 32 rows x 8 chunks

    const __bf16* gA0 = A + (m0 + arow) * (long)K + achk * 8;
    const __bf16* gA1 = A + (m0 + arow + 64) * (long)K + achk * 8;
    const __bf16* gB  = Wt + (long)brow * N + n0 + bchk * 8;
    __bf16* lA0 = &sA[arow * 40 + achk * 8];
    __bf16* lA1 = &sA[(arow + 64) * 40 + achk * 8];
    __bf16* lB  = &sB[brow * 72 + bchk * 8];
    const long wstep = (long)32 * N;

    // prologue: first tile into registers
    u32x4 ra0 = *(const u32x4*)gA0;
    u32x4 ra1 = *(const u32x4*)gA1;
    u32x4 rb  = *(const u32x4*)gB;

    v8f acc[2][2] = {{{}, {}}, {{}, {}}};

    for (int k0 = 0; k0 < K; k0 += 32) {
        __syncthreads();
        *(u32x4*)lA0 = ra0;
        *(u32x4*)lA1 = ra1;
        *(u32x4*)lB  = rb;
        if (k0 + 32 < K) {           // prefetch next tile; overlaps WMMA below
            gA0 += 32; gA1 += 32; gB += wstep;
            ra0 = *(const u32x4*)gA0;
            ra1 = *(const u32x4*)gA1;
            rb  = *(const u32x4*)gB;
        }
        __syncthreads();

        // A fragments: row = lane&15, K chunks {half*8, half*8+16}
        Frag16 ua[2], ub[2];
#pragma unroll
        for (int mi = 0; mi < 2; ++mi) {
            const int ar = mw * 32 + mi * 16 + r16;
            ua[mi].q[0] = *(const u32x4*)&sA[ar * 40 + half * 8];
            ua[mi].q[1] = *(const u32x4*)&sA[ar * 40 + half * 8 + 16];
        }
        // B fragments: col = lane&15, K rows {half*8.., half*8+16..}
#pragma unroll
        for (int ni = 0; ni < 2; ++ni) {
            const int bc = nw * 32 + ni * 16 + r16;
#pragma unroll
            for (int s = 0; s < 8; ++s) {
                ub[ni].v[s]     = sB[(half * 8 + s) * 72 + bc];
                ub[ni].v[s + 8] = sB[(half * 8 + 16 + s) * 72 + bc];
            }
        }
#pragma unroll
        for (int mi = 0; mi < 2; ++mi)
#pragma unroll
            for (int ni = 0; ni < 2; ++ni)
                acc[mi][ni] = __builtin_amdgcn_wmma_f32_16x16x32_bf16(
                    false, ua[mi].v, false, ub[ni].v, (short)0, acc[mi][ni], false, false);
    }

    // C layout: VGPR r -> M = r + half*8 ; N = lane&15
#pragma unroll
    for (int mi = 0; mi < 2; ++mi) {
        const long mrow = m0 + mw * 32 + mi * 16 + half * 8;
#pragma unroll
        for (int ni = 0; ni < 2; ++ni) {
            const long ncol = n0 + nw * 32 + ni * 16 + r16;
#pragma unroll
            for (int r = 0; r < 8; ++r) {
                const long row = mrow + r;
                float v = acc[mi][ni][r];
                if (EPI == 0) {
                    outF[row * N + ncol] = v;
                } else if (EPI == 1) {
                    outH[row * N + ncol] = (__bf16)(v * sigf(v));
                } else {
                    outF[row * N + ncol] = v + bias[ncol];
                }
            }
        }
    }
}

// ---------------- depthwise causal conv (tap=4) + SiLU ----------------
__global__ void k_conv_silu(const float* __restrict__ xz, const float* __restrict__ cw,
                            const float* __restrict__ cb, float* __restrict__ xcf,
                            __bf16* __restrict__ xcb) {
    int i = blockIdx.x * 256 + threadIdx.x;
    if (i >= BL * DINNER) return;
    int d  = i & (DINNER - 1);
    int bl = i >> 9;                 // token index
    int l  = bl & (SEQL - 1);
    int b  = bl >> 11;
    float acc = cb[d];
#pragma unroll
    for (int k = 0; k < 4; ++k) {
        int t = l - 3 + k;
        if (t >= 0) acc += xz[((long)(b * SEQL + t)) * 1024 + d] * cw[d * 4 + k];
    }
    float s = acc * sigf(acc);
    xcf[i] = s;
    xcb[i] = (__bf16)s;
}

// ---------------- delta = softplus(dt @ dt_proj_w^T + b) ----------------
__global__ void k_delta(const float* __restrict__ xdbl, const float* __restrict__ dtw,
                        const float* __restrict__ dtb, float* __restrict__ delta) {
    int i = blockIdx.x * 256 + threadIdx.x;
    if (i >= BL * DINNER) return;
    int d  = i & (DINNER - 1);
    int bl = i >> 9;
    const float* row = xdbl + (long)bl * XDBL_P;
    float acc = dtb[d];
#pragma unroll
    for (int r = 0; r < DTRANK; ++r) acc += row[r] * dtw[d * DTRANK + r];
    delta[i] = (acc > 20.0f) ? acc : log1pf(__expf(acc));
}

// ---------------- selective scan: chunked linear-recurrence ----------------
// pass 1: per (b,d,chunk) partial state with h0=0, plus S = sum(delta)
__global__ __launch_bounds__(512)
void k_scan1(const float* __restrict__ delta, const float* __restrict__ xcf,
             const float* __restrict__ xdbl, const float* __restrict__ A_log,
             float* __restrict__ hp, float* __restrict__ Ssum) {
    const int b = blockIdx.x / NCHUNK;
    const int c = blockIdx.x % NCHUNK;
    const int d = threadIdx.x;
    float An[DSTATE], h[DSTATE];
#pragma unroll
    for (int n = 0; n < DSTATE; ++n) { An[n] = -__expf(A_log[d * DSTATE + n]); h[n] = 0.0f; }
    float S = 0.0f;
    const int t0 = c * CHUNK;
    for (int t = 0; t < CHUNK; ++t) {
        const long idx = (long)(b * SEQL + t0 + t);
        const float dlt = delta[idx * DINNER + d];
        const float xv  = xcf[idx * DINNER + d];
        const float bx  = dlt * xv;
        const float* row = xdbl + idx * XDBL_P;
        S += dlt;
#pragma unroll
        for (int n = 0; n < DSTATE; ++n)
            h[n] = __expf(dlt * An[n]) * h[n] + bx * row[DTRANK + n];
    }
    const long base = (((long)(b * DINNER + d)) * NCHUNK + c) * DSTATE;
#pragma unroll
    for (int n = 0; n < DSTATE; ++n) hp[base + n] = h[n];
    Ssum[(long)(b * DINNER + d) * NCHUNK + c] = S;
}

// pass 2: carry states across chunks (decay over chunk = exp(A * S))
__global__ void k_scan2(const float* __restrict__ hp, const float* __restrict__ Ssum,
                        const float* __restrict__ A_log, float* __restrict__ h0b) {
    int i = blockIdx.x * 256 + threadIdx.x;
    if (i >= BATCH * DINNER) return;
    const int d = i & (DINNER - 1);
    float An[DSTATE], h[DSTATE];
#pragma unroll
    for (int n = 0; n < DSTATE; ++n) { An[n] = -__expf(A_log[d * DSTATE + n]); h[n] = 0.0f; }
    for (int c = 0; c < NCHUNK; ++c) {
        const long base = ((long)i * NCHUNK + c) * DSTATE;
#pragma unroll
        for (int n = 0; n < DSTATE; ++n) h0b[base + n] = h[n];
        const float S = Ssum[(long)i * NCHUNK + c];
#pragma unroll
        for (int n = 0; n < DSTATE; ++n) h[n] = __expf(An[n] * S) * h[n] + hp[base + n];
    }
}

// pass 3: replay with correct h0, fuse y = C.h + D*x, gate with silu(z), emit bf16
__global__ __launch_bounds__(512)
void k_scan3(const float* __restrict__ delta, const float* __restrict__ xcf,
             const float* __restrict__ xdbl, const float* __restrict__ A_log,
             const float* __restrict__ h0b, const float* __restrict__ Dskip,
             const float* __restrict__ xz, __bf16* __restrict__ ybf) {
    const int b = blockIdx.x / NCHUNK;
    const int c = blockIdx.x % NCHUNK;
    const int d = threadIdx.x;
    float An[DSTATE], h[DSTATE];
    const long hb = (((long)(b * DINNER + d)) * NCHUNK + c) * DSTATE;
#pragma unroll
    for (int n = 0; n < DSTATE; ++n) {
        An[n] = -__expf(A_log[d * DSTATE + n]);
        h[n]  = h0b[hb + n];
    }
    const float dsk = Dskip[d];
    const int t0 = c * CHUNK;
    for (int t = 0; t < CHUNK; ++t) {
        const long idx = (long)(b * SEQL + t0 + t);
        const float dlt = delta[idx * DINNER + d];
        const float xv  = xcf[idx * DINNER + d];
        const float bx  = dlt * xv;
        const float* row = xdbl + idx * XDBL_P;
        float y = 0.0f;
#pragma unroll
        for (int n = 0; n < DSTATE; ++n) {
            h[n] = __expf(dlt * An[n]) * h[n] + bx * row[DTRANK + n];
            y += h[n] * row[DTRANK + DSTATE + n];
        }
        y += xv * dsk;
        const float z = xz[idx * 1024 + DINNER + d];
        y *= z * sigf(z);
        ybf[idx * DINNER + d] = (__bf16)y;
    }
}

// ---------------- LayerNorm over (L, NOUT) per batch ----------------
__global__ __launch_bounds__(256)
void k_red1(const float* __restrict__ x, float2* __restrict__ part) {
    const int b   = blockIdx.x >> 7;
    const int blk = blockIdx.x & 127;
    const long base = (long)b * SEQL * NOUT + (long)blk * 4096;
    float s = 0.0f, q = 0.0f;
    for (int i = threadIdx.x; i < 4096; i += 256) {
        float v = x[base + i];
        s += v; q += v * v;
    }
    __shared__ float ss[256], sq[256];
    ss[threadIdx.x] = s; sq[threadIdx.x] = q;
    __syncthreads();
    for (int st = 128; st > 0; st >>= 1) {
        if (threadIdx.x < st) {
            ss[threadIdx.x] += ss[threadIdx.x + st];
            sq[threadIdx.x] += sq[threadIdx.x + st];
        }
        __syncthreads();
    }
    if (threadIdx.x == 0) part[blockIdx.x] = make_float2(ss[0], sq[0]);
}

__global__ __launch_bounds__(128)
void k_red2(const float2* __restrict__ part, float2* __restrict__ musig) {
    const int b = blockIdx.x;
    __shared__ float ss[128], sq[128];
    float2 p = part[b * 128 + threadIdx.x];
    ss[threadIdx.x] = p.x; sq[threadIdx.x] = p.y;
    __syncthreads();
    for (int st = 64; st > 0; st >>= 1) {
        if (threadIdx.x < st) {
            ss[threadIdx.x] += ss[threadIdx.x + st];
            sq[threadIdx.x] += sq[threadIdx.x + st];
        }
        __syncthreads();
    }
    if (threadIdx.x == 0) {
        const float cnt = (float)(SEQL * NOUT);
        float mu  = ss[0] / cnt;
        float var = sq[0] / cnt - mu * mu;
        musig[b] = make_float2(mu, rsqrtf(var + LN_EPS));
    }
}

__global__ void k_final(const float* __restrict__ x, const float2* __restrict__ musig,
                        const float* __restrict__ lnw, const float* __restrict__ lnb,
                        float* __restrict__ out) {
    int i = blockIdx.x * 256 + threadIdx.x;
    if (i >= BATCH * SEQL * NOUT) return;
    const int b  = i / (SEQL * NOUT);
    const int lo = i % (SEQL * NOUT);
    const float2 ms = musig[b];
    out[i] = (x[i] - ms.x) * ms.y * lnw[lo] + lnb[lo];
}

// ---------------- launcher ----------------
extern "C" void kernel_launch(void* const* d_in, const int* in_sizes, int n_in,
                              void* d_out, int out_size, void* d_ws, size_t ws_size,
                              hipStream_t stream) {
    const float* inputs   = (const float*)d_in[0];
    const float* inprojw  = (const float*)d_in[1];
    const float* convw    = (const float*)d_in[2];
    const float* convb    = (const float*)d_in[3];
    const float* xprojw   = (const float*)d_in[4];
    const float* dtprojw  = (const float*)d_in[5];
    const float* dtprojb  = (const float*)d_in[6];
    const float* A_log    = (const float*)d_in[7];
    const float* Dskip    = (const float*)d_in[8];
    const float* outprojw = (const float*)d_in[9];
    const float* dimw     = (const float*)d_in[10];
    const float* dimb     = (const float*)d_in[11];
    const float* lnw      = (const float*)d_in[12];
    const float* lnb      = (const float*)d_in[13];
    float* out = (float*)d_out;

    char* base = (char*)d_ws;
    size_t off = 0;
    auto alloc = [&](size_t bytes) -> void* {
        void* p = base + off;
        off = (off + bytes + 255) & ~(size_t)255;
        return p;
    };
    __bf16* in_bf  = (__bf16*)alloc((size_t)BL * DMODEL * 2);
    __bf16* wA     = (__bf16*)alloc((size_t)DMODEL * 1024 * 2);           // (256,1024)
    __bf16* wX     = (__bf16*)alloc((size_t)DINNER * XDBL_P * 2);         // (512,64)
    __bf16* wO     = (__bf16*)alloc((size_t)DINNER * DMODEL * 2);         // (512,256)
    __bf16* wD     = (__bf16*)alloc((size_t)DMODEL * NOUT * 2);           // (256,256)
    float*  xz     = (float*)alloc((size_t)BL * 1024 * 4);
    float*  xcf    = (float*)alloc((size_t)BL * DINNER * 4);
    __bf16* xcb    = (__bf16*)alloc((size_t)BL * DINNER * 2);
    float*  xdbl   = (float*)alloc((size_t)BL * XDBL_P * 4);
    float*  delta  = (float*)alloc((size_t)BL * DINNER * 4);
    float*  Ssum   = (float*)alloc((size_t)BATCH * DINNER * NCHUNK * 4);
    float*  hp     = (float*)alloc((size_t)BATCH * DINNER * NCHUNK * DSTATE * 4);
    float*  h0b    = (float*)alloc((size_t)BATCH * DINNER * NCHUNK * DSTATE * 4);
    __bf16* ybf    = (__bf16*)alloc((size_t)BL * DINNER * 2);
    float2* part   = (float2*)alloc((size_t)BATCH * 128 * sizeof(float2));
    float2* musig  = (float2*)alloc((size_t)BATCH * sizeof(float2));
    // safe aliases (producer strictly after last consumer of the aliased buffer)
    __bf16* sbf    = xcb;           // silu(out_proj) bf16; xcb dead after x_proj GEMM
    float*  dimout = xz;            // xz dead after scan3

    const int B256 = 256;
    auto gs = [](long n, int b) { return (unsigned)((n + b - 1) / b); };

    // 1) convert activations, transpose+convert weights to (K,N) bf16
    k_cvt_f32_bf16<<<gs((long)BL * DMODEL, B256), B256, 0, stream>>>(inputs, in_bf, BL * DMODEL);
    k_transpose_cvt<<<gs((long)DMODEL * 1024, B256), B256, 0, stream>>>(inprojw, wA, 1024, DMODEL, 1024);
    k_transpose_cvt<<<gs((long)DINNER * XDBL_P, B256), B256, 0, stream>>>(xprojw, wX, 48, DINNER, XDBL_P);
    k_transpose_cvt<<<gs((long)DINNER * DMODEL, B256), B256, 0, stream>>>(outprojw, wO, DMODEL, DINNER, DMODEL);
    k_transpose_cvt<<<gs((long)DMODEL * NOUT, B256), B256, 0, stream>>>(dimw, wD, NOUT, DMODEL, NOUT);

    // 2) in_proj: xz = u @ W^T   (32768 x 1024, K=256)
    gemm_bf16<0><<<dim3(BL / 128, 1024 / 64), 256, 0, stream>>>(in_bf, wA, xz, nullptr, nullptr,
                                                                BL, 1024, DMODEL);
    // 3) depthwise conv + silu
    k_conv_silu<<<gs((long)BL * DINNER, B256), B256, 0, stream>>>(xz, convw, convb, xcf, xcb);
    // 4) x_proj: x_dbl = xc @ W^T  (32768 x 64pad, K=512)
    gemm_bf16<0><<<dim3(BL / 128, XDBL_P / 64), 256, 0, stream>>>(xcb, wX, xdbl, nullptr, nullptr,
                                                                  BL, XDBL_P, DINNER);
    // 5) delta = softplus(dt @ dt_proj^T + b)
    k_delta<<<gs((long)BL * DINNER, B256), B256, 0, stream>>>(xdbl, dtprojw, dtprojb, delta);
    // 6) chunked selective scan
    k_scan1<<<BATCH * NCHUNK, DINNER, 0, stream>>>(delta, xcf, xdbl, A_log, hp, Ssum);
    k_scan2<<<gs((long)BATCH * DINNER, B256), B256, 0, stream>>>(hp, Ssum, A_log, h0b);
    k_scan3<<<BATCH * NCHUNK, DINNER, 0, stream>>>(delta, xcf, xdbl, A_log, h0b, Dskip, xz, ybf);
    // 7) out_proj + silu -> bf16   (32768 x 256, K=512)
    gemm_bf16<1><<<dim3(BL / 128, DMODEL / 64), 256, 0, stream>>>(ybf, wO, nullptr, sbf, nullptr,
                                                                  BL, DMODEL, DINNER);
    // 8) dim proj + bias -> f32    (32768 x 256, K=256)
    gemm_bf16<2><<<dim3(BL / 128, NOUT / 64), 256, 0, stream>>>(sbf, wD, dimout, nullptr, dimb,
                                                                BL, NOUT, DMODEL);
    // 9) LayerNorm over (L, NOUT) per batch (deterministic tree reduction)
    k_red1<<<BATCH * 128, 256, 0, stream>>>(dimout, part);
    k_red2<<<BATCH, 128, 0, stream>>>(part, musig);
    k_final<<<gs((long)BATCH * SEQL * NOUT, B256), B256, 0, stream>>>(dimout, musig, lnw, lnb, out);
}
